// STHG_42210938585358
// MI455X (gfx1250) — matrix operand order repaired and checked by reading, not compiled, not run on hardware
//
#include <hip/hip_runtime.h>

// ---------------------------------------------------------------------------
// MI455X (gfx1250) implementation of the heterogeneous-GNN reference.
// All dense math runs on v_wmma_f32_16x16x32_bf16 (bf16 in, f32 acc).
// GEMM LDS tiles are staged by the Tensor Data Mover with double-buffered
// ping-pong (tensor_load_to_lds + s_wait_tensorcnt 1) so DMA overlaps WMMA.
// Manual global->LDS staging is the fallback. Irregular graph ops use atomics.
// ---------------------------------------------------------------------------

typedef __bf16 bf16;
typedef __attribute__((ext_vector_type(16))) __bf16 v16bf;
typedef __attribute__((ext_vector_type(8)))  __bf16 v8bf;
typedef __attribute__((ext_vector_type(8)))  float  v8f;
typedef __attribute__((ext_vector_type(4)))  unsigned int v4u;
typedef __attribute__((ext_vector_type(8)))  int v8i;
typedef __attribute__((ext_vector_type(4)))  int v4i;

#define LDT 40   // LDS row stride in bf16 elements (32 data + 8 pad)

#if __has_builtin(__builtin_amdgcn_tensor_load_to_lds) && __has_builtin(__builtin_amdgcn_s_wait_tensorcnt)
#define STHG_HAS_TDM 1
#else
#define STHG_HAS_TDM 0
#endif

__device__ __forceinline__ v16bf sthg_ldpair(const bf16* p0, const bf16* p1) {
  union { v16bf v; v8bf h[2]; } u;
  u.h[0] = *(const v8bf*)p0;
  u.h[1] = *(const v8bf*)p1;
  return u.v;
}

#if STHG_HAS_TDM
__device__ __forceinline__ unsigned sthg_lds_off(const void* p) {
  return (unsigned)(uintptr_t)(__attribute__((address_space(3))) const void*)p;
}

// 2D TDM load: tile [tile_d1 rows x tile_d0 elems] of bf16 from row-major
// global (row stride = stride0 elems) into LDS with a 4-DWORD pad inserted
// after every 16 DWORDs (i.e., LDS row stride = 32 data + 8 pad bf16 = LDT).
// Rows >= tensor_d1 are zero-filled by TDM OOB handling.
__device__ __forceinline__ void sthg_tdm_load_2d(
    unsigned lds_off, const void* gptr, unsigned tile_d0, unsigned tile_d1,
    unsigned tensor_d0, unsigned tensor_d1, unsigned long long stride0) {
  unsigned long long ga = (unsigned long long)(uintptr_t)gptr;
  v4u g0;
  g0[0] = 1u;                                        // count=1 (valid user D#)
  g0[1] = lds_off;                                   // LDS byte address
  g0[2] = (unsigned)(ga & 0xFFFFFFFFu);              // global_addr[31:0]
  g0[3] = (unsigned)((ga >> 32) & 0x1FFFFFFu)        // global_addr[56:32]
        | (2u << 30);                                // type=2 ("image")
  v8i g1;
  g1[0] = (int)((1u << 16)                           // data_size = 2 bytes
              | (1u << 20)                           // pad_enable
              | (3u << 22)                           // pad_interval = 16 DW
              | (3u << 25));                         // pad_amount  = 4 DW
  g1[1] = (int)((tensor_d0 & 0xFFFFu) << 16);        // tensor_dim0[15:0]
  g1[2] = (int)(((tensor_d0 >> 16) & 0xFFFFu)        // tensor_dim0[31:16]
              | ((tensor_d1 & 0xFFFFu) << 16));      // tensor_dim1[15:0]
  g1[3] = (int)(((tensor_d1 >> 16) & 0xFFFFu)        // tensor_dim1[31:16]
              | ((tile_d0 & 0xFFFFu) << 16));        // tile_dim0
  g1[4] = (int)(tile_d1 & 0xFFFFu);                  // tile_dim1 (tile_dim2=0)
  g1[5] = (int)(unsigned)(stride0 & 0xFFFFFFFFull);  // dim0_stride[31:0]
  g1[6] = (int)(unsigned)((stride0 >> 32) & 0xFFFFull); // dim0_stride[47:32]
  g1[7] = 0;                                         // dim1_stride = 0 (2D)
  v4i g2 = {0, 0, 0, 0};
  v4i g3 = {0, 0, 0, 0};
#if defined(__clang_major__) && (__clang_major__ >= 23)
  v8i g4 = {0, 0, 0, 0, 0, 0, 0, 0};
  __builtin_amdgcn_tensor_load_to_lds(g0, g1, g2, g3, g4, 0);
#else
  __builtin_amdgcn_tensor_load_to_lds(g0, g1, g2, g3, 0);
#endif
}
#endif  // STHG_HAS_TDM

// ---------------------------------------------------------------------------
// WMMA GEMM: C[M,N] = A[M,K] @ B[K,N]. A bf16 row-major (stride K),
// Bt = B^T bf16 row-major [N,K] (stride K). Epilogue: +bias, relu,
// accumulate into f32 C, optional bf16 secondary output.
// Block: 128x128 tile, 256 threads = 8 waves, each wave 32x64 via 2x4 WMMA.
// N must be a multiple of 128 (all layer widths here are 128/512/1024).
// TDM path: double-buffered LDS tiles; wave0/wave1 issue the next K-slab's
// tensor_load_to_lds while all 8 waves run WMMA on the current slab.
// ---------------------------------------------------------------------------
__global__ __launch_bounds__(256) void sthg_gemm(
    const bf16* __restrict__ A, const bf16* __restrict__ Bt,
    const float* __restrict__ bias, float* __restrict__ Cf,
    bf16* __restrict__ Cbf, int M, int N, int K, int accum, int dorelu) {
#if STHG_HAS_TDM
  __shared__ __attribute__((aligned(16))) bf16 As[2][128 * LDT];
  __shared__ __attribute__((aligned(16))) bf16 Bs[2][128 * LDT];
#else
  __shared__ __attribute__((aligned(16))) bf16 As[1][128 * LDT];
  __shared__ __attribute__((aligned(16))) bf16 Bs[1][128 * LDT];
#endif
  const int tid  = threadIdx.x;
  const int wave = tid >> 5;
  const int lane = tid & 31;
  const int m0 = blockIdx.y * 128;
  const int n0 = blockIdx.x * 128;
  const int wm = (wave >> 1) * 32;   // 0,32,64,96
  const int wn = (wave & 1) * 64;    // 0,64

  v8f acc[2][4];
#pragma unroll
  for (int i = 0; i < 2; ++i)
#pragma unroll
    for (int j = 0; j < 4; ++j)
#pragma unroll
      for (int r = 0; r < 8; ++r) acc[i][j][r] = 0.0f;

#if STHG_HAS_TDM
  const unsigned remM = (unsigned)(M - m0);
  // Prologue: issue DMA for the first K-slab into buffer 0.
  if (wave == 0) {
    sthg_tdm_load_2d(sthg_lds_off(As[0]), A + (size_t)m0 * K, 32u, 128u,
                     32u, remM, (unsigned long long)K);
  } else if (wave == 1) {
    sthg_tdm_load_2d(sthg_lds_off(Bs[0]), Bt + (size_t)n0 * K, 32u, 128u,
                     32u, 128u, (unsigned long long)K);
  }
  int t = 0;
#else
  const int lr = tid >> 1;          // 0..127: tile row staged by this thread
  const int lc = (tid & 1) * 16;    // 0 or 16 halfs within the K slab
  v8bf zero8;
#pragma unroll
  for (int q = 0; q < 8; ++q) zero8[q] = (bf16)0.0f;
#endif

  for (int k0 = 0; k0 < K; k0 += 32) {
    __syncthreads();   // all waves done reading the buffer we are about to fill
#if STHG_HAS_TDM
    const bool hasnext = (k0 + 32 < K);
    if (wave == 0) {
      if (hasnext) {
        sthg_tdm_load_2d(sthg_lds_off(As[t ^ 1]), A + (size_t)m0 * K + k0 + 32,
                         32u, 128u, 32u, remM, (unsigned long long)K);
        __builtin_amdgcn_s_wait_tensorcnt(1);  // in-order: current slab landed
      } else {
        __builtin_amdgcn_s_wait_tensorcnt(0);
      }
    } else if (wave == 1) {
      if (hasnext) {
        sthg_tdm_load_2d(sthg_lds_off(Bs[t ^ 1]), Bt + (size_t)n0 * K + k0 + 32,
                         32u, 128u, 32u, 128u, (unsigned long long)K);
        __builtin_amdgcn_s_wait_tensorcnt(1);
      } else {
        __builtin_amdgcn_s_wait_tensorcnt(0);
      }
    }
    const bf16* Ab = As[t];
    const bf16* Bb = Bs[t];
#else
    {
      const int gr = m0 + lr;
      v8bf a0 = zero8, a1 = zero8;
      if (gr < M) {
        const bf16* pa = A + (size_t)gr * K + k0 + lc;
        a0 = *(const v8bf*)pa;
        a1 = *(const v8bf*)(pa + 8);
        if (k0 + 32 < K) __builtin_prefetch(pa + 32, 0, 1);  // global_prefetch_b8
      }
      *(v8bf*)&As[0][lr * LDT + lc]     = a0;
      *(v8bf*)&As[0][lr * LDT + lc + 8] = a1;
      const bf16* pb = Bt + (size_t)(n0 + lr) * K + k0 + lc;  // N mult of 128
      *(v8bf*)&Bs[0][lr * LDT + lc]     = *(const v8bf*)pb;
      *(v8bf*)&Bs[0][lr * LDT + lc + 8] = *(const v8bf*)(pb + 8);
      if (k0 + 32 < K) __builtin_prefetch(pb + 32, 0, 1);
    }
    const bf16* Ab = As[0];
    const bf16* Bb = Bs[0];
#endif
    __syncthreads();

    // A-frag: ISA layout (lanes<16: K 0-7 & 16-23; lanes>=16: K 8-15 & 24-31)
    // B-frag: same ISA layout (lanes<16: K 0-7 & 16-23; lanes>=16: K 8-15 & 24-31)
    const int ah = (lane < 16) ? 0 : 8;
    const int bh = (lane < 16) ? 0 : 8;
    v16bf af[2], bf_[4];
#pragma unroll
    for (int i = 0; i < 2; ++i) {
      const int lm = wm + i * 16 + (lane & 15);
      af[i] = sthg_ldpair(&Ab[lm * LDT + ah], &Ab[lm * LDT + ah + 16]);
    }
#pragma unroll
    for (int j = 0; j < 4; ++j) {
      const int ln = wn + j * 16 + (lane & 15);
      bf_[j] = sthg_ldpair(&Bb[ln * LDT + bh], &Bb[ln * LDT + bh + 16]);
    }
#pragma unroll
    for (int i = 0; i < 2; ++i)
#pragma unroll
      for (int j = 0; j < 4; ++j)
        acc[i][j] = __builtin_amdgcn_wmma_f32_16x16x32_bf16(
            false, af[i], false, bf_[j], (short)0, acc[i][j], false, false);
#if STHG_HAS_TDM
    t ^= 1;
#endif
  }

  // Epilogue per documented C/D layout: VGPR r -> m = base + r + 8*(lane>=16)
#pragma unroll
  for (int i = 0; i < 2; ++i) {
#pragma unroll
    for (int j = 0; j < 4; ++j) {
      const int n = n0 + wn + j * 16 + (lane & 15);
#pragma unroll
      for (int r = 0; r < 8; ++r) {
        const int m = m0 + wm + i * 16 + r + ((lane >> 4) << 3);
        if (m < M) {
          float v = acc[i][j][r];
          if (bias)   v += bias[n];
          if (dorelu) v = fmaxf(v, 0.0f);
          const size_t off = (size_t)m * N + n;
          if (Cf)  { if (accum) v += Cf[off]; Cf[off] = v; }
          if (Cbf) Cbf[off] = (bf16)v;
        }
      }
    }
  }
}

// ---------------------------------------------------------------------------
// Elementwise / reduction / graph kernels
// ---------------------------------------------------------------------------
__global__ void sthg_fill(float* p, float v, size_t n) {
  size_t i = (size_t)blockIdx.x * 256 + threadIdx.x;
  if (i < n) p[i] = v;
}

__global__ void sthg_cvt_bf16(const float* __restrict__ x, bf16* __restrict__ y, size_t n) {
  size_t i = (size_t)blockIdx.x * 256 + threadIdx.x;
  if (i < n) y[i] = (bf16)x[i];
}

// W f32 [K,N] -> Wt bf16 [N,K]
__global__ void sthg_xpose(const float* __restrict__ W, bf16* __restrict__ Wt, int K, int N) {
  size_t i = (size_t)blockIdx.x * 256 + threadIdx.x;
  if (i < (size_t)K * N) {
    int k = (int)(i / N), n = (int)(i % N);
    Wt[(size_t)n * K + k] = (bf16)W[i];
  }
}

// Per-column mean/var (biased) over M rows. grid.x = N, block = 256.
__global__ void sthg_colstats(const float* __restrict__ X, float* __restrict__ stats,
                              int M, int N) {
  const int j = blockIdx.x;
  __shared__ float ssum[256], ssq[256];
  float s = 0.f, q = 0.f;
  for (int i = threadIdx.x; i < M; i += 256) {
    float v = X[(size_t)i * N + j];
    s += v; q += v * v;
  }
  ssum[threadIdx.x] = s; ssq[threadIdx.x] = q;
  __syncthreads();
  for (int st = 128; st > 0; st >>= 1) {
    if (threadIdx.x < st) { ssum[threadIdx.x] += ssum[threadIdx.x + st];
                            ssq[threadIdx.x]  += ssq[threadIdx.x + st]; }
    __syncthreads();
  }
  if (threadIdx.x == 0) {
    float mean = ssum[0] / (float)M;
    stats[j]     = mean;
    stats[N + j] = ssq[0] / (float)M - mean * mean;
  }
}

// In-place BN(train stats) + ReLU, also emits bf16 copy.
__global__ void sthg_bn_relu(float* __restrict__ X, bf16* __restrict__ Xbf,
                             const float* __restrict__ stats,
                             const float* __restrict__ g, const float* __restrict__ b,
                             int N, size_t total) {
  size_t i = (size_t)blockIdx.x * 256 + threadIdx.x;
  if (i >= total) return;
  int j = (int)(i % N);
  float y = (X[i] - stats[j]) * rsqrtf(stats[N + j] + 1e-5f) * g[j] + b[j];
  y = fmaxf(y, 0.0f);
  X[i] = y;
  Xbf[i] = (bf16)y;
}

// EdgeConv input gather: out[e, 0:C] = x[dst], out[e, C:2C] = x[src]-x[dst]
__global__ void sthg_edge_gather(const float* __restrict__ X,
                                 const int* __restrict__ src, const int* __restrict__ dst,
                                 int ecnt, int C, bf16* __restrict__ out) {
  size_t total = (size_t)ecnt * 2 * C;
  size_t i = (size_t)blockIdx.x * 256 + threadIdx.x;
  if (i >= total) return;
  const int C2 = 2 * C;
  int e = (int)(i / C2), f = (int)(i % C2);
  float xi = X[(size_t)dst[e] * C + (f < C ? f : f - C)];
  float v  = (f < C) ? xi : (X[(size_t)src[e] * C + (f - C)] - xi);
  out[i] = (bf16)v;
}

__device__ __forceinline__ void sthg_atomicMaxF(float* addr, float val) {
  int* ai = (int*)addr;
  int old = __float_as_int(*addr);
  while (__int_as_float(old) < val) {
    int prev = atomicCAS(ai, old, __float_as_int(val));
    if (prev == old) break;
    old = prev;
  }
}

// mode: 0 -> attr<=0, 1 -> attr>=0, 2 -> always
__global__ void sthg_scatter_max(const float* __restrict__ msg, const int* __restrict__ dst,
                                 const float* __restrict__ attr, int mode,
                                 float* __restrict__ out, int ecnt, int N) {
  size_t total = (size_t)ecnt * N;
  size_t i = (size_t)blockIdx.x * 256 + threadIdx.x;
  if (i >= total) return;
  int e = (int)(i / N), n = (int)(i % N);
  bool ok = (mode == 2) || (mode == 0 ? (attr[e] <= 0.f) : (attr[e] >= 0.f));
  if (ok) sthg_atomicMaxF(&out[(size_t)dst[e] * N + n], msg[i]);
}

__global__ void sthg_max_finalize(float* __restrict__ X, size_t n) {
  size_t i = (size_t)blockIdx.x * 256 + threadIdx.x;
  if (i < n) { float v = X[i]; X[i] = (v > -5e29f) ? v : 0.0f; }
}

__global__ void sthg_seg_sum(const float* __restrict__ X,
                             const int* __restrict__ src, const int* __restrict__ dst,
                             const float* __restrict__ attr, int mode,
                             float* __restrict__ sum, float* __restrict__ cnt,
                             int E, int C) {
  size_t total = (size_t)E * C;
  size_t i = (size_t)blockIdx.x * 256 + threadIdx.x;
  if (i >= total) return;
  int e = (int)(i / C), f = (int)(i % C);
  bool ok = (mode == 2) || (mode == 0 ? (attr[e] <= 0.f) : (attr[e] >= 0.f));
  if (ok) {
    atomicAdd(&sum[(size_t)dst[e] * C + f], X[(size_t)src[e] * C + f]);
    if (f == 0) atomicAdd(&cnt[dst[e]], 1.0f);
  }
}

__global__ void sthg_mean_bf16(const float* __restrict__ sum, const float* __restrict__ cnt,
                               bf16* __restrict__ out, int Nd, int C) {
  size_t total = (size_t)Nd * C;
  size_t i = (size_t)blockIdx.x * 256 + threadIdx.x;
  if (i >= total) return;
  float c = cnt[i / C];
  if (c < 1.0f) c = 1.0f;
  out[i] = (bf16)(sum[i] / c);
}

// ---------------------------------------------------------------------------
// Host orchestration
// ---------------------------------------------------------------------------
extern "C" void kernel_launch(void* const* d_in, const int* in_sizes, int n_in,
                              void* d_out, int out_size, void* d_ws, size_t ws_size,
                              hipStream_t stream) {
  (void)in_sizes; (void)n_in; (void)out_size; (void)ws_size;
  enum { NO_ = 20000, NT_ = 2000, DIN_ = 1536, TD_ = 1024, C0_ = 512, C1_ = 1024,
         FD_ = 128, EOO_ = 160000, ETO_ = 80000, ETT_ = 16000, ECHUNK_ = 16384 };
  const float NEGF = -1e30f;

  char* base = (char*)d_ws;
  size_t cur = 0;
  auto alloc = [&](size_t bytes) -> void* {
    void* p = base + cur;
    cur = (cur + bytes + 255) & ~(size_t)255;
    return p;
  };

  // --- inputs (setup_inputs dict order; params flattened in insertion order)
  const float* x_o = (const float*)d_in[0];
  const float* x_t = (const float*)d_in[1];
  const int* oo_src = (const int*)d_in[2]; const int* oo_dst = oo_src + EOO_;
  const int* to_src = (const int*)d_in[3]; const int* to_dst = to_src + ETO_;
  const int* tt_src = (const int*)d_in[4]; const int* tt_dst = tt_src + ETT_;
  const float* oo_attr = (const float*)d_in[5];
  const float* tt_attr = (const float*)d_in[6];
  auto P = [&](int i) { return (const float*)d_in[i]; };
  // param leaf index map (insertion order):
  // 7 l011.W  8 l011.b  9 bn01.g 10 bn01.b  11 l01t.W 12 l01t.b 13 bn01t.g 14 bn01t.b
  // 15..25 l11 {ooW1,oob1,ooW2,oob2,toWL,tobL,toWR,ttW1,ttb1,ttW2,ttb2}, 26..36 l12, 37..47 l13
  // 48..59 bn11/bn11t/bn12/bn12t/bn13/bn13t {g,b}
  // 60..68 l21 {ooWL,oobL,ooWR,toWL,tobL,toWR,ttWL,ttbL,ttWR}
  // 69..72 bn21{g,b} bn21t{g,b}
  // 73..81 l31 {ooWL,oobL,ooWR,toWL,tobL,toWR,ttWL,ttbL,ttWR}, 82..90 l32, 91..99 l33
  const int L1B[3] = {15, 26, 37};
  const int L3B[3] = {73, 82, 91};

  // --- workspace
  bf16*  xo_in_bf = (bf16*)alloc((size_t)NO_ * DIN_ * 2);
  bf16*  xt_in_bf = (bf16*)alloc((size_t)NT_ * TD_ * 2);
  float* xo   = (float*)alloc((size_t)NO_ * C0_ * 4);
  bf16*  xo_bf = (bf16*)alloc((size_t)NO_ * C0_ * 2);
  float* xt   = (float*)alloc((size_t)NT_ * TD_ * 4);
  bf16*  xt_bf = (bf16*)alloc((size_t)NT_ * TD_ * 2);
  float* o1   = (float*)alloc((size_t)NO_ * C0_ * 4);
  bf16*  o1_bf = (bf16*)alloc((size_t)NO_ * C0_ * 2);
  float* t1   = (float*)alloc((size_t)NT_ * TD_ * 4);
  bf16*  t1_bf = (bf16*)alloc((size_t)NT_ * TD_ * 2);
  float* o2   = (float*)alloc((size_t)NO_ * C1_ * 4);
  bf16*  o2_bf = (bf16*)alloc((size_t)NO_ * C1_ * 2);
  float* t2   = (float*)alloc((size_t)NT_ * C1_ * 4);
  bf16*  t2_bf = (bf16*)alloc((size_t)NT_ * C1_ * 2);
  float* agg  = (float*)alloc((size_t)NO_ * C1_ * 4);
  float* cnt  = (float*)alloc((size_t)NO_ * 4);
  bf16*  meanbf = (bf16*)alloc((size_t)NO_ * C1_ * 2);
  float* stats = (float*)alloc((size_t)2 * 1024 * 4);
  bf16*  ein  = (bf16*)alloc((size_t)ECHUNK_ * 2048 * 2);
  bf16*  ehid = (bf16*)alloc((size_t)ECHUNK_ * 1024 * 2);
  float* emsg = (float*)alloc((size_t)ECHUNK_ * 1024 * 4);

  auto xpose = [&](int idx, int K, int N) -> bf16* {
    bf16* p = (bf16*)alloc((size_t)K * N * 2);
    size_t tot = (size_t)K * N;
    sthg_xpose<<<(unsigned)((tot + 255) / 256), 256, 0, stream>>>(P(idx), p, K, N);
    return p;
  };

  // --- transpose + bf16-convert every GEMM weight: Wt [N,K]
  bf16* wt011 = xpose(7, DIN_, C0_);
  bf16* wt01t = xpose(11, TD_, TD_);
  bf16 *wooW1[3], *wooW2[3], *wtoWL[3], *wtoWR[3], *wttW1[3], *wttW2[3];
  for (int b = 0; b < 3; ++b) {
    int q = L1B[b];
    wooW1[b] = xpose(q + 0, 2 * C0_, C0_);
    wooW2[b] = xpose(q + 2, C0_, C0_);
    wtoWL[b] = xpose(q + 4, TD_, C0_);
    wtoWR[b] = xpose(q + 6, C0_, C0_);
    wttW1[b] = xpose(q + 7, 2 * TD_, TD_);
    wttW2[b] = xpose(q + 9, TD_, TD_);
  }
  bf16* w21ooWL = xpose(60, C0_, C1_);
  bf16* w21ooWR = xpose(62, C0_, C1_);
  bf16* w21toWL = xpose(63, TD_, C1_);
  bf16* w21toWR = xpose(65, C0_, C1_);
  bf16* w21ttWL = xpose(66, TD_, C1_);
  bf16* w21ttWR = xpose(68, TD_, C1_);
  bf16 *w3ooWL[3], *w3ooWR[3], *w3toWL[3], *w3toWR[3];
  for (int b = 0; b < 3; ++b) {
    int q = L3B[b];
    w3ooWL[b] = xpose(q + 0, C1_, FD_);
    w3ooWR[b] = xpose(q + 2, C1_, FD_);
    w3toWL[b] = xpose(q + 3, C1_, FD_);
    w3toWR[b] = xpose(q + 5, C1_, FD_);
  }

  // --- helpers
  auto fillf = [&](float* p, float v, size_t n) {
    sthg_fill<<<(unsigned)((n + 255) / 256), 256, 0, stream>>>(p, v, n);
  };
  auto gemm = [&](const bf16* A, const bf16* Bt, const float* bias, float* Cf, bf16* Cbf,
                  int M, int N, int K, int accum, int relu) {
    dim3 g((N + 127) / 128, (M + 127) / 128);
    sthg_gemm<<<g, 256, 0, stream>>>(A, Bt, bias, Cf, Cbf, M, N, K, accum, relu);
  };
  auto bnrelu = [&](float* X, bf16* Xbf, int gi, int bi, int M, int N) {
    sthg_colstats<<<N, 256, 0, stream>>>(X, stats, M, N);
    size_t tot = (size_t)M * N;
    sthg_bn_relu<<<(unsigned)((tot + 255) / 256), 256, 0, stream>>>(X, Xbf, stats, P(gi), P(bi), N, tot);
  };
  auto segmean = [&](const float* Xs, const int* es, const int* ed, const float* attr, int mode,
                     int E, int C, int Nd) {
    fillf(agg, 0.f, (size_t)Nd * C);
    fillf(cnt, 0.f, (size_t)Nd);
    size_t tot = (size_t)E * C;
    sthg_seg_sum<<<(unsigned)((tot + 255) / 256), 256, 0, stream>>>(Xs, es, ed, attr, mode, agg, cnt, E, C);
    size_t tn = (size_t)Nd * C;
    sthg_mean_bf16<<<(unsigned)((tn + 255) / 256), 256, 0, stream>>>(agg, cnt, meanbf, Nd, C);
  };
  auto edgeconv = [&](const float* Xf, const int* es, const int* ed, const float* attr, int mode,
                      int E, int C, int Cout, const bf16* W1t, const float* b1,
                      const bf16* W2t, const float* b2, float* outb, int Nn) {
    const int Ch = Cout;  // mlp(2C, C, C): hidden == output width for both oo/tt
    fillf(outb, NEGF, (size_t)Nn * Cout);
    for (int e0 = 0; e0 < E; e0 += ECHUNK_) {
      int ec = (E - e0 < ECHUNK_) ? (E - e0) : ECHUNK_;
      size_t tg = (size_t)ec * 2 * C;
      sthg_edge_gather<<<(unsigned)((tg + 255) / 256), 256, 0, stream>>>(Xf, es + e0, ed + e0, ec, C, ein);
      gemm(ein, W1t, b1, nullptr, ehid, ec, Ch, 2 * C, 0, 1);     // relu, bf16 out
      gemm(ehid, W2t, b2, emsg, nullptr, ec, Cout, Ch, 0, 0);
      size_t ts = (size_t)ec * Cout;
      sthg_scatter_max<<<(unsigned)((ts + 255) / 256), 256, 0, stream>>>(
          emsg, ed + e0, attr ? attr + e0 : nullptr, mode, outb, ec, Cout);
    }
    size_t tn = (size_t)Nn * Cout;
    sthg_max_finalize<<<(unsigned)((tn + 255) / 256), 256, 0, stream>>>(outb, tn);
  };

  // --- stage 0: input projections + BN + ReLU
  { size_t n = (size_t)NO_ * DIN_;
    sthg_cvt_bf16<<<(unsigned)((n + 255) / 256), 256, 0, stream>>>(x_o, xo_in_bf, n); }
  { size_t n = (size_t)NT_ * TD_;
    sthg_cvt_bf16<<<(unsigned)((n + 255) / 256), 256, 0, stream>>>(x_t, xt_in_bf, n); }
  gemm(xo_in_bf, wt011, P(8), xo, nullptr, NO_, C0_, DIN_, 0, 0);
  bnrelu(xo, xo_bf, 9, 10, NO_, C0_);
  gemm(xt_in_bf, wt01t, P(12), xt, nullptr, NT_, TD_, TD_, 0, 0);
  bnrelu(xt, xt_bf, 13, 14, NT_, TD_);

  fillf((float*)d_out, 0.f, (size_t)NO_ * FD_);

  // --- three branches (forward / backward / all edge masks), summed into d_out
  for (int br = 0; br < 3; ++br) {
    const int mode = (br == 0) ? 0 : ((br == 1) ? 1 : 2);
    const int q = L1B[br];

    // o1 = EdgeConv_oo(xo) + SAGE_to(xt -> xo); then BN+ReLU
    edgeconv(xo, oo_src, oo_dst, oo_attr, mode, EOO_, C0_, C0_,
             wooW1[br], P(q + 1), wooW2[br], P(q + 3), o1, NO_);
    segmean(xt, to_src, to_dst, nullptr, 2, ETO_, TD_, NO_);
    gemm(meanbf, wtoWL[br], P(q + 5), o1, nullptr, NO_, C0_, TD_, 1, 0);
    gemm(xo_bf, wtoWR[br], nullptr, o1, nullptr, NO_, C0_, C0_, 1, 0);
    bnrelu(o1, o1_bf, 48 + 4 * br, 49 + 4 * br, NO_, C0_);

    // t1 = EdgeConv_tt(xt); BN+ReLU
    edgeconv(xt, tt_src, tt_dst, tt_attr, mode, ETT_, TD_, TD_,
             wttW1[br], P(q + 8), wttW2[br], P(q + 10), t1, NT_);
    bnrelu(t1, t1_bf, 50 + 4 * br, 51 + 4 * br, NT_, TD_);

    // stage 2 (shared l21)
    segmean(o1, oo_src, oo_dst, oo_attr, mode, EOO_, C0_, NO_);
    gemm(meanbf, w21ooWL, P(61), o2, nullptr, NO_, C1_, C0_, 0, 0);
    gemm(o1_bf, w21ooWR, nullptr, o2, nullptr, NO_, C1_, C0_, 1, 0);
    segmean(t1, to_src, to_dst, nullptr, 2, ETO_, TD_, NO_);
    gemm(meanbf, w21toWL, P(64), o2, nullptr, NO_, C1_, TD_, 1, 0);
    gemm(o1_bf, w21toWR, nullptr, o2, nullptr, NO_, C1_, C0_, 1, 0);
    segmean(t1, tt_src, tt_dst, tt_attr, mode, ETT_, TD_, NT_);
    gemm(meanbf, w21ttWL, P(67), t2, nullptr, NT_, C1_, TD_, 0, 0);
    gemm(t1_bf, w21ttWR, nullptr, t2, nullptr, NT_, C1_, TD_, 1, 0);
    bnrelu(o2, o2_bf, 69, 70, NO_, C1_);
    bnrelu(t2, t2_bf, 71, 72, NT_, C1_);

    // stage 3 -> accumulate into d_out
    const int q3 = L3B[br];
    segmean(o2, oo_src, oo_dst, oo_attr, mode, EOO_, C1_, NO_);
    gemm(meanbf, w3ooWL[br], P(q3 + 1), (float*)d_out, nullptr, NO_, FD_, C1_, 1, 0);
    gemm(o2_bf, w3ooWR[br], nullptr, (float*)d_out, nullptr, NO_, FD_, C1_, 1, 0);
    segmean(t2, to_src, to_dst, nullptr, 2, ETO_, C1_, NO_);
    gemm(meanbf, w3toWL[br], P(q3 + 4), (float*)d_out, nullptr, NO_, FD_, C1_, 1, 0);
    gemm(o2_bf, w3toWR[br], nullptr, (float*)d_out, nullptr, NO_, FD_, C1_, 1, 0);
  }
}
